// ResnetBase_77077483094593
// MI455X (gfx1250) — matrix operand order repaired and checked by reading, not compiled, not run on hardware
//
#include <hip/hip_runtime.h>

#define BATCH 16384
#define FDIM  512
#define HDIM  256
#define ODIM  32
#define NEXP  8

// LDS row strides (padded to avoid bank conflicts on ds_load_b128 A-fragments)
#define XS (FDIM + 8)   // 520 ushorts per row
#define HS (HDIM + 8)   // 264 ushorts per row

typedef __attribute__((ext_vector_type(16))) __bf16 v16bf;
typedef __attribute__((ext_vector_type(8)))  float  v8f;

union BfFrag { v16bf v; uint4 q[2]; };

__device__ __forceinline__ unsigned short f2bfbits(float f) {
  unsigned u = __builtin_bit_cast(unsigned, f);
  u += 0x7FFFu + ((u >> 16) & 1u);          // round-to-nearest-even
  return (unsigned short)(u >> 16);
}

// ---------------- pass 0: zero expert counters ----------------
__global__ void k_zero(int* counts) {
  if (threadIdx.x < 32) counts[threadIdx.x] = 0;
}

// ---------------- pass 1: bucket rows by expert ----------------
__global__ void k_bucket(const int* __restrict__ act, int* __restrict__ counts,
                         int* __restrict__ rowlist) {
  int i = blockIdx.x * blockDim.x + threadIdx.x;
  if (i < BATCH) {
    int e = act[i] & 7;
    int p = atomicAdd(&counts[e], 1);
    rowlist[e * BATCH + p] = i;
  }
}

// ------- pass 2: weights fp32 [e][k][n] -> bf16 transposed [e][n][k] -------
__global__ void k_prep(const float* __restrict__ W1, const float* __restrict__ W2,
                       const float* __restrict__ W3,
                       unsigned short* __restrict__ W1t, unsigned short* __restrict__ W2t,
                       unsigned short* __restrict__ W3t) {
  int tid = blockIdx.x * blockDim.x + threadIdx.x;
  int nthreads = gridDim.x * blockDim.x;
  for (int idx = tid; idx < NEXP * FDIM * HDIM; idx += nthreads) {
    int e = idx / (FDIM * HDIM);
    int r = idx - e * FDIM * HDIM;
    int n = r / FDIM;
    int k = r - n * FDIM;
    W1t[idx] = f2bfbits(W1[(e * FDIM + k) * HDIM + n]);
  }
  for (int idx = tid; idx < NEXP * HDIM * HDIM; idx += nthreads) {
    int e = idx / (HDIM * HDIM);
    int r = idx - e * HDIM * HDIM;
    int n = r / HDIM;
    int k = r - n * HDIM;
    W2t[idx] = f2bfbits(W2[(e * HDIM + k) * HDIM + n]);
  }
  for (int idx = tid; idx < NEXP * ODIM * HDIM; idx += nthreads) {
    int e = idx / (ODIM * HDIM);
    int r = idx - e * ODIM * HDIM;
    int n = r / HDIM;
    int k = r - n * HDIM;
    W3t[idx] = f2bfbits(W3[(e * HDIM + k) * ODIM + n]);
  }
}

// ---------------- pass 3: tiled bf16-WMMA MoE MLP ----------------
__global__ __launch_bounds__(128) void k_moe(
    const float* __restrict__ feat,
    const float* __restrict__ b1, const float* __restrict__ b2,
    const float* __restrict__ b3,
    const unsigned short* __restrict__ W1t, const unsigned short* __restrict__ W2t,
    const unsigned short* __restrict__ W3t,
    const int* __restrict__ rowlist, const int* __restrict__ counts,
    float* __restrict__ out) {
  __shared__ __align__(16) unsigned short Xs[16 * XS];
  __shared__ __align__(16) unsigned short H1s[16 * HS];
  __shared__ __align__(16) unsigned short H2s[16 * HS];
  __shared__ int rows[16];

  const int e = blockIdx.y;
  const int t = blockIdx.x;
  const int cnt = counts[e];
  if (t * 16 >= cnt) return;   // block-uniform early exit (over-launched tiles)

  const int tid  = threadIdx.x;
  const int lane = tid & 31;
  const int wave = tid >> 5;
  const int hh   = lane >> 4;  // half-wave select
  const int nn   = lane & 15;  // tile column (B/C/D), also A-row index

  if (tid < 16) {
    int i = t * 16 + tid;
    rows[tid] = (i < cnt) ? rowlist[e * BATCH + i] : -1;
  }
  __syncthreads();

  // gather 16 rows of features, convert to bf16 into LDS (float4 loads)
  for (int ii = tid; ii < 16 * (FDIM / 4); ii += 128) {
    int m  = ii >> 7;               // FDIM/4 == 128 quads per row
    int k4 = (ii & 127) << 2;
    int row = rows[m];
    float4 v = make_float4(0.f, 0.f, 0.f, 0.f);
    if (row >= 0) v = *(const float4*)(feat + row * FDIM + k4);
    uint2 p;
    p.x = (unsigned)f2bfbits(v.x) | ((unsigned)f2bfbits(v.y) << 16);
    p.y = (unsigned)f2bfbits(v.z) | ((unsigned)f2bfbits(v.w) << 16);
    *(uint2*)(&Xs[m * XS + k4]) = p;
  }
  __syncthreads();

  // ---------------- layer 1: (16x512) @ (512x256) + b1, ReLU ----------------
  for (int ntl = 0; ntl < 4; ++ntl) {
    const int ncol = (wave * 4 + ntl) * 16 + nn;
    const float bv = b1[e * HDIM + ncol];
    v8f acc;
#pragma unroll
    for (int r = 0; r < 8; ++r) acc[r] = bv;
    const unsigned short* wrow = W1t + (e * HDIM + ncol) * FDIM;
#pragma unroll
    for (int kk = 0; kk < FDIM / 32; ++kk) {
      const int k0 = kk * 32;
      BfFrag a, b;
      a.q[0] = *(const uint4*)(&Xs[nn * XS + k0 + hh * 8]);
      a.q[1] = *(const uint4*)(&Xs[nn * XS + k0 + 16 + hh * 8]);
      b.q[0] = *(const uint4*)(wrow + k0 + hh * 8);
      b.q[1] = *(const uint4*)(wrow + k0 + 16 + hh * 8);
      acc = __builtin_amdgcn_wmma_f32_16x16x32_bf16(false, a.v, false, b.v,
                                                    (short)0, acc, false, false);
    }
#pragma unroll
    for (int r = 0; r < 8; ++r) {
      float x = acc[r] > 0.f ? acc[r] : 0.f;
      H1s[(r + 8 * hh) * HS + ncol] = f2bfbits(x);
    }
  }
  __syncthreads();

  // ---------------- layer 2: (16x256) @ (256x256) + b2, ReLU ----------------
  for (int ntl = 0; ntl < 4; ++ntl) {
    const int ncol = (wave * 4 + ntl) * 16 + nn;
    const float bv = b2[e * HDIM + ncol];
    v8f acc;
#pragma unroll
    for (int r = 0; r < 8; ++r) acc[r] = bv;
    const unsigned short* wrow = W2t + (e * HDIM + ncol) * HDIM;
#pragma unroll
    for (int kk = 0; kk < HDIM / 32; ++kk) {
      const int k0 = kk * 32;
      BfFrag a, b;
      a.q[0] = *(const uint4*)(&H1s[nn * HS + k0 + hh * 8]);
      a.q[1] = *(const uint4*)(&H1s[nn * HS + k0 + 16 + hh * 8]);
      b.q[0] = *(const uint4*)(wrow + k0 + hh * 8);
      b.q[1] = *(const uint4*)(wrow + k0 + 16 + hh * 8);
      acc = __builtin_amdgcn_wmma_f32_16x16x32_bf16(false, a.v, false, b.v,
                                                    (short)0, acc, false, false);
    }
#pragma unroll
    for (int r = 0; r < 8; ++r) {
      float x = acc[r] > 0.f ? acc[r] : 0.f;
      H2s[(r + 8 * hh) * HS + ncol] = f2bfbits(x);
    }
  }
  __syncthreads();

  // ------- layer 3: (16x256) @ (256x16 slice of W3) + b3, keep cols 0..7 -------
  if (wave == 0) {            // wave-uniform branch: EXEC stays all-1s for WMMA
    const int ncol = nn;      // output columns 0..15 (only 0..7 stored)
    const float bv = b3[e * ODIM + ncol];
    v8f acc;
#pragma unroll
    for (int r = 0; r < 8; ++r) acc[r] = bv;
    const unsigned short* wrow = W3t + (e * ODIM + ncol) * HDIM;
#pragma unroll
    for (int kk = 0; kk < HDIM / 32; ++kk) {
      const int k0 = kk * 32;
      BfFrag a, b;
      a.q[0] = *(const uint4*)(&H2s[nn * HS + k0 + hh * 8]);
      a.q[1] = *(const uint4*)(&H2s[nn * HS + k0 + 16 + hh * 8]);
      b.q[0] = *(const uint4*)(wrow + k0 + hh * 8);
      b.q[1] = *(const uint4*)(wrow + k0 + 16 + hh * 8);
      acc = __builtin_amdgcn_wmma_f32_16x16x32_bf16(false, a.v, false, b.v,
                                                    (short)0, acc, false, false);
    }
#pragma unroll
    for (int r = 0; r < 8; ++r) {
      int m = r + 8 * hh;
      int row = rows[m];
      if (row >= 0 && nn < 8) out[row * 8 + nn] = acc[r];
    }
  }
}

extern "C" void kernel_launch(void* const* d_in, const int* in_sizes, int n_in,
                              void* d_out, int out_size, void* d_ws, size_t ws_size,
                              hipStream_t stream) {
  const float* feat = (const float*)d_in[0];
  const float* W1   = (const float*)d_in[1];
  const float* b1   = (const float*)d_in[2];
  const float* W2   = (const float*)d_in[3];
  const float* b2   = (const float*)d_in[4];
  const float* W3   = (const float*)d_in[5];
  const float* b3   = (const float*)d_in[6];
  const int*   act  = (const int*)d_in[7];

  char* ws = (char*)d_ws;
  int* counts  = (int*)ws;                                   // 128 B
  int* rowlist = (int*)(ws + 256);                           // 8*16384*4 = 512 KB
  unsigned short* W1t = (unsigned short*)(ws + 256 + NEXP * BATCH * 4);
  unsigned short* W2t = W1t + NEXP * HDIM * FDIM;            // +2 MB
  unsigned short* W3t = W2t + NEXP * HDIM * HDIM;            // +1 MB (total ~3.8 MB)

  k_zero<<<1, 32, 0, stream>>>(counts);
  k_bucket<<<(BATCH + 255) / 256, 256, 0, stream>>>(act, counts, rowlist);
  k_prep<<<512, 256, 0, stream>>>(W1, W2, W3, W1t, W2t, W3t);

  dim3 grid(BATCH / 16, NEXP);
  k_moe<<<grid, 128, 0, stream>>>(feat, b1, b2, b3, W1t, W2t, W3t,
                                  rowlist, counts, (float*)d_out);
}
